// MLPEdgeEncoder_44890998178542
// MI455X (gfx1250) — matrix operand order repaired and checked by reading, not compiled, not run on hardware
//
#include <hip/hip_runtime.h>
#include <hip/hip_bf16.h>

// ---------------------------------------------------------------------------
// NRI MLPEdgeEncoder for MI455X (gfx1250, wave32, WMMA f32_16x16x32_f16)
//
// B=64, V=64, E=4032, D=H=128.  All GEMMs done with v_wmma_f32_16x16x32_f16
// (f16 inputs, f32 accumulate).  BatchNorm folded into per-feature affine
// (scale/shift) applied at the consumer of each stage.  Activations stored
// as f16 (halves HBM traffic).  Stats computed via deterministic per-block
// partials + fixed-order reduction (no float atomics anywhere).
// ---------------------------------------------------------------------------

typedef __attribute__((ext_vector_type(16))) _Float16 v16h;
typedef __attribute__((ext_vector_type(8)))  _Float16 v8h;
typedef __attribute__((ext_vector_type(8)))  float    v8f;

#define NRI_B 64
#define NRI_V 64
#define NRI_E 4032
#define NRI_H 128

struct StageArgs {
    const float*    xf32;     // MODE 0 input (row-major f32, width = KIN)
    const _Float16* srcA;     // f16 activation source (h_act / n_in / n_act)
    const _Float16* srcSkip;  // MODE 3: skip connection (e_act)
    const float*    scA;      // affine for srcA   (nullptr-like unused ok)
    const float*    shA;
    const float*    scS;      // affine for srcSkip
    const float*    shS;
    const int*      send;
    const int*      recv;
    const _Float16* wp1;      // packed W1 fragments
    const _Float16* wp2;      // packed W2 fragments
    const float*    b1;
    const float*    b2;
    _Float16*       actOut;   // [rows][128] f16, post-ELU, pre-BN
    float*          partial;  // [nBlocks][256] : 128 sums then 128 sumsqs
};

// Pack weight matrix W (KIN x 128, row-major f32) into WMMA B-fragment order:
//   packed[((kt*8 + ct)*32 + lane)*16 + e] = W[kt*32 + (e%8) + 8*(lane>=16) + 16*(e>=8)][ct*16 + lane%16]
__global__ __launch_bounds__(256) void pack_w_kernel(const float* __restrict__ W,
                                                     _Float16* __restrict__ out,
                                                     int kin) {
    int idx = blockIdx.x * 256 + threadIdx.x;
    if (idx >= kin * 128) return;
    int kt   = idx >> 12;          // 8 ct * 32 lane * 16 e = 4096 per k-step
    int rem  = idx & 4095;
    int ct   = rem >> 9;
    int rem2 = rem & 511;
    int lane = rem2 >> 4;
    int e    = rem2 & 15;
    int K    = kt * 32 + (e & 7) + ((lane >> 4) << 3) + ((e >> 3) << 4);
    int col  = ct * 16 + (lane & 15);
    out[idx] = (_Float16)W[K * 128 + col];
}

// Fused two-layer MLP stage with WMMA.
// MODE 0: read f32 rows directly (stage 1, KIN=128)
// MODE 1: gather [h(send) | h(recv)] with affine (stage 2, KIN=256)
// MODE 2: read f16 rows directly, no affine (stage 3, KIN=128)
// MODE 3: gather [n(send) | n(recv) | e_skip] with affines (stage 4, KIN=384)
template <int KIN, int MODE>
__global__ __launch_bounds__(256) void mlp_stage_kernel(StageArgs args) {
    constexpr int KP  = KIN + 8;   // padded LDS row stride (halves); +16B kills bank conflicts
    constexpr int KP2 = NRI_H + 8;

    __shared__ _Float16 Xt[64 * KP];
    __shared__ _Float16 Tt[64 * KP2];
    __shared__ float pS[128 * 8];
    __shared__ float pQ[128 * 8];

    const int tid   = threadIdx.x;
    const int lane  = tid & 31;
    const int wave  = tid >> 5;
    const int rs    = wave & 3;   // 16-row strip within the 64-row tile
    const int ch    = wave >> 2;  // 64-column half of the 128 outputs
    const long rowBase = (long)blockIdx.x * 64;

    // ---- phase 0: build f16 input tile in LDS -----------------------------
    for (int idx = tid; idx < 64 * KIN; idx += 256) {
        int r = idx / KIN;
        int c = idx - r * KIN;
        long grow = rowBase + r;
        float v;
        if constexpr (MODE == 0) {
            v = args.xf32[grow * KIN + c];
        } else if constexpr (MODE == 2) {
            v = (float)args.srcA[grow * NRI_H + c];
        } else if constexpr (MODE == 1) {
            int b = (int)(grow / NRI_E);
            int e = (int)(grow - (long)b * NRI_E);
            int f = c & 127;
            int node = (c < 128) ? args.send[e] : args.recv[e];
            float hv = (float)args.srcA[((long)(b * NRI_V + node)) * NRI_H + f];
            v = args.scA[f] * hv + args.shA[f];
        } else {  // MODE == 3
            int b = (int)(grow / NRI_E);
            int e = (int)(grow - (long)b * NRI_E);
            if (c < 256) {
                int f = c & 127;
                int node = (c < 128) ? args.send[e] : args.recv[e];
                float hv = (float)args.srcA[((long)(b * NRI_V + node)) * NRI_H + f];
                v = args.scA[f] * hv + args.shA[f];
            } else {
                int f = c - 256;
                float hv = (float)args.srcSkip[grow * NRI_H + f];
                v = args.scS[f] * hv + args.shS[f];
            }
        }
        Xt[r * KP + c] = (_Float16)v;
    }
    __syncthreads();

    const int rowL  = rs * 16 + (lane & 15);
    const int hiOff = (lane >> 4) << 3;  // 0 / 8: A-frag K-interleave (ISA 7.12.2)
    const int rb    = rs * 16 + hiOff;   // C/D row base for this lane half

    // ---- GEMM 1: T = elu(X @ W1 + b1) -------------------------------------
    {
        v8f acc[4] = {};
        for (int kt = 0; kt < KIN / 32; ++kt) {
            const _Float16* ap = &Xt[rowL * KP + kt * 32 + hiOff];
            v8h alo = *(const v8h*)(ap);
            v8h ahi = *(const v8h*)(ap + 16);
            v16h a = __builtin_shufflevector(alo, ahi, 0, 1, 2, 3, 4, 5, 6, 7,
                                             8, 9, 10, 11, 12, 13, 14, 15);
#pragma unroll
            for (int ct = 0; ct < 4; ++ct) {
                v16h bfrag = *(const v16h*)&args.wp1[(((kt * 8) + (ch * 4 + ct)) * 32 + lane) * 16];
                acc[ct] = __builtin_amdgcn_wmma_f32_16x16x32_f16(
                    false, a, false, bfrag, (short)0, acc[ct], false, false);
            }
        }
#pragma unroll
        for (int ct = 0; ct < 4; ++ct) {
            int col = ch * 64 + ct * 16 + (lane & 15);
            float bias = args.b1[col];
#pragma unroll
            for (int i = 0; i < 8; ++i) {
                float v = acc[ct][i] + bias;
                v = v > 0.f ? v : expm1f(v);  // ELU(alpha=1)
                Tt[(rb + i) * KP2 + col] = (_Float16)v;
            }
        }
    }
    __syncthreads();

    // ---- GEMM 2: Y = elu(T @ W2 + b2); store f16 + BN partial stats -------
    {
        v8f acc[4] = {};
        for (int kt = 0; kt < NRI_H / 32; ++kt) {
            const _Float16* ap = &Tt[rowL * KP2 + kt * 32 + hiOff];
            v8h alo = *(const v8h*)(ap);
            v8h ahi = *(const v8h*)(ap + 16);
            v16h a = __builtin_shufflevector(alo, ahi, 0, 1, 2, 3, 4, 5, 6, 7,
                                             8, 9, 10, 11, 12, 13, 14, 15);
#pragma unroll
            for (int ct = 0; ct < 4; ++ct) {
                v16h bfrag = *(const v16h*)&args.wp2[(((kt * 8) + (ch * 4 + ct)) * 32 + lane) * 16];
                acc[ct] = __builtin_amdgcn_wmma_f32_16x16x32_f16(
                    false, a, false, bfrag, (short)0, acc[ct], false, false);
            }
        }
        const int slot = rs * 2 + (lane >> 4);  // 8 deterministic slots per column
#pragma unroll
        for (int ct = 0; ct < 4; ++ct) {
            int col = ch * 64 + ct * 16 + (lane & 15);
            float bias = args.b2[col];
            float s = 0.f, q = 0.f;
#pragma unroll
            for (int i = 0; i < 8; ++i) {
                float v = acc[ct][i] + bias;
                v = v > 0.f ? v : expm1f(v);
                args.actOut[(rowBase + rb + i) * NRI_H + col] = (_Float16)v;
                s += v;
                q += v * v;
            }
            pS[col * 8 + slot] = s;
            pQ[col * 8 + slot] = q;
        }
    }
    __syncthreads();

    if (tid < 128) {  // fixed-order in-block reduction -> global partials
        float s = 0.f, q = 0.f;
#pragma unroll
        for (int k = 0; k < 8; ++k) { s += pS[tid * 8 + k]; q += pQ[tid * 8 + k]; }
        args.partial[(long)blockIdx.x * 256 + tid]       = s;
        args.partial[(long)blockIdx.x * 256 + 128 + tid] = q;
    }
}

// Fixed-order reduction of per-block partials -> BN affine (scale/shift).
__global__ __launch_bounds__(128) void bn_finalize_kernel(const float* __restrict__ partial,
                                                          int nBlocks, float invN,
                                                          const float* __restrict__ gamma,
                                                          const float* __restrict__ beta,
                                                          float* __restrict__ scale,
                                                          float* __restrict__ shift) {
    int f = threadIdx.x;
    float s = 0.f, q = 0.f;
    for (int i = 0; i < nBlocks; ++i) {
        s += partial[(long)i * 256 + f];
        q += partial[(long)i * 256 + 128 + f];
    }
    float m   = s * invN;
    float var = q * invN - m * m;           // biased variance, matches torch BN
    float sc  = gamma[f] * rsqrtf(var + 1e-5f);
    scale[f] = sc;
    shift[f] = beta[f] - m * sc;
}

// edge2node: n[b,v,f] = (1/V) * sum_{t<63} BN2(e[b, v*63+t, f])
// (edges from nonzero(ones-eye) are receiver-grouped: recv(e) == e/63)
__global__ __launch_bounds__(128) void edge2node_kernel(const _Float16* __restrict__ eAct,
                                                        const float* __restrict__ sc,
                                                        const float* __restrict__ sh,
                                                        _Float16* __restrict__ nIn) {
    int bv = blockIdx.x;          // b*V + v
    int f  = threadIdx.x;
    int b  = bv >> 6;
    int v  = bv & 63;
    long rowStart = (long)b * NRI_E + (long)v * 63;
    float s = 0.f;
    for (int t = 0; t < 63; ++t)
        s += (float)eAct[(rowStart + t) * NRI_H + f];
    float val = (sc[f] * s + 63.0f * sh[f]) * (1.0f / (float)NRI_V);
    nIn[(long)bv * NRI_H + f] = (_Float16)val;
}

// out[row] = sigmoid(BN4(e2[row]) . fc_W + fc_b), one wave per row.
__global__ __launch_bounds__(256) void fc_sigmoid_kernel(const _Float16* __restrict__ e2,
                                                         const float* __restrict__ sc,
                                                         const float* __restrict__ sh,
                                                         const float* __restrict__ fcW,
                                                         const float* __restrict__ fcb,
                                                         float* __restrict__ out) {
    int wave = threadIdx.x >> 5;
    int lane = threadIdx.x & 31;
    long row = (long)blockIdx.x * 8 + wave;
    const _Float16* p = e2 + row * NRI_H + lane * 4;
    float acc = 0.f;
#pragma unroll
    for (int j = 0; j < 4; ++j) {
        int f = lane * 4 + j;
        float v = sc[f] * (float)p[j] + sh[f];
        acc += v * fcW[f];
    }
#pragma unroll
    for (int o = 16; o > 0; o >>= 1) acc += __shfl_xor(acc, o, 32);
    if (lane == 0) out[row] = 1.0f / (1.0f + expf(-(acc + fcb[0])));
}

// ---------------------------------------------------------------------------
extern "C" void kernel_launch(void* const* d_in, const int* in_sizes, int n_in,
                              void* d_out, int out_size, void* d_ws, size_t ws_size,
                              hipStream_t stream) {
    (void)in_sizes; (void)n_in; (void)out_size; (void)ws_size;

    // Inputs in setup_inputs() order (tuples flattened in order):
    const float* x    = (const float*)d_in[0];
    const float* W1_1 = (const float*)d_in[1],  *b1_1 = (const float*)d_in[2];
    const float* W2_1 = (const float*)d_in[3],  *b2_1 = (const float*)d_in[4];
    const float* g_1  = (const float*)d_in[5],  *be_1 = (const float*)d_in[6];
    const float* W1_2 = (const float*)d_in[7],  *b1_2 = (const float*)d_in[8];
    const float* W2_2 = (const float*)d_in[9],  *b2_2 = (const float*)d_in[10];
    const float* g_2  = (const float*)d_in[11], *be_2 = (const float*)d_in[12];
    const float* W1_3 = (const float*)d_in[13], *b1_3 = (const float*)d_in[14];
    const float* W2_3 = (const float*)d_in[15], *b2_3 = (const float*)d_in[16];
    const float* g_3  = (const float*)d_in[17], *be_3 = (const float*)d_in[18];
    const float* W1_4 = (const float*)d_in[19], *b1_4 = (const float*)d_in[20];
    const float* W2_4 = (const float*)d_in[21], *b2_4 = (const float*)d_in[22];
    const float* g_4  = (const float*)d_in[23], *be_4 = (const float*)d_in[24];
    const float* fcW  = (const float*)d_in[25];
    const float* fcb  = (const float*)d_in[26];
    const int* recv   = (const int*)d_in[27];
    const int* send   = (const int*)d_in[28];
    float* out = (float*)d_out;

    // ---- workspace carving (aligned 256B) ---------------------------------
    size_t off = 0;
    auto carve = [&](size_t bytes) -> void* {
        off = (off + 255) & ~(size_t)255;
        void* p = (char*)d_ws + off;
        off += bytes;
        return p;
    };
    _Float16* pk1w1 = (_Float16*)carve(128 * 128 * 2);
    _Float16* pk1w2 = (_Float16*)carve(128 * 128 * 2);
    _Float16* pk2w1 = (_Float16*)carve(256 * 128 * 2);
    _Float16* pk2w2 = (_Float16*)carve(128 * 128 * 2);
    _Float16* pk3w1 = (_Float16*)carve(128 * 128 * 2);
    _Float16* pk3w2 = (_Float16*)carve(128 * 128 * 2);
    _Float16* pk4w1 = (_Float16*)carve(384 * 128 * 2);
    _Float16* pk4w2 = (_Float16*)carve(128 * 128 * 2);
    float* partial  = (float*)carve((size_t)4032 * 256 * 4);
    float* scale1 = (float*)carve(512); float* shift1 = (float*)carve(512);
    float* scale2 = (float*)carve(512); float* shift2 = (float*)carve(512);
    float* scale3 = (float*)carve(512); float* shift3 = (float*)carve(512);
    float* scale4 = (float*)carve(512); float* shift4 = (float*)carve(512);
    _Float16* h_act    = (_Float16*)carve((size_t)NRI_B * NRI_V * NRI_H * 2);
    _Float16* e_act    = (_Float16*)carve((size_t)NRI_B * NRI_E * NRI_H * 2);
    _Float16* n_in_act = (_Float16*)carve((size_t)NRI_B * NRI_V * NRI_H * 2);
    _Float16* n_act    = (_Float16*)carve((size_t)NRI_B * NRI_V * NRI_H * 2);
    _Float16* e2_act   = (_Float16*)carve((size_t)NRI_B * NRI_E * NRI_H * 2);

    // ---- 0) pack all weights to WMMA fragment order -----------------------
    pack_w_kernel<<<(128 * 128 + 255) / 256, 256, 0, stream>>>(W1_1, pk1w1, 128);
    pack_w_kernel<<<(128 * 128 + 255) / 256, 256, 0, stream>>>(W2_1, pk1w2, 128);
    pack_w_kernel<<<(256 * 128 + 255) / 256, 256, 0, stream>>>(W1_2, pk2w1, 256);
    pack_w_kernel<<<(128 * 128 + 255) / 256, 256, 0, stream>>>(W2_2, pk2w2, 128);
    pack_w_kernel<<<(128 * 128 + 255) / 256, 256, 0, stream>>>(W1_3, pk3w1, 128);
    pack_w_kernel<<<(128 * 128 + 255) / 256, 256, 0, stream>>>(W2_3, pk3w2, 128);
    pack_w_kernel<<<(384 * 128 + 255) / 256, 256, 0, stream>>>(W1_4, pk4w1, 384);
    pack_w_kernel<<<(128 * 128 + 255) / 256, 256, 0, stream>>>(W2_4, pk4w2, 128);

    const int nodeRows = NRI_B * NRI_V;         // 4096
    const long edgeRows = (long)NRI_B * NRI_E;  // 258048

    // ---- 1) mlp1 over nodes ----------------------------------------------
    {
        StageArgs a = {};
        a.xf32 = x; a.wp1 = pk1w1; a.wp2 = pk1w2; a.b1 = b1_1; a.b2 = b2_1;
        a.actOut = h_act; a.partial = partial;
        mlp_stage_kernel<128, 0><<<nodeRows / 64, 256, 0, stream>>>(a);
        bn_finalize_kernel<<<1, 128, 0, stream>>>(partial, nodeRows / 64,
                                                  1.0f / nodeRows, g_1, be_1, scale1, shift1);
    }
    // ---- 2) mlp2 over edges (gather send|recv of BN1(h)) ------------------
    {
        StageArgs a = {};
        a.srcA = h_act; a.scA = scale1; a.shA = shift1;
        a.send = send; a.recv = recv;
        a.wp1 = pk2w1; a.wp2 = pk2w2; a.b1 = b1_2; a.b2 = b2_2;
        a.actOut = e_act; a.partial = partial;
        mlp_stage_kernel<256, 1><<<(int)(edgeRows / 64), 256, 0, stream>>>(a);
        bn_finalize_kernel<<<1, 128, 0, stream>>>(partial, (int)(edgeRows / 64),
                                                  1.0f / (float)edgeRows, g_2, be_2, scale2, shift2);
    }
    // ---- 3) edge2node segmented sum, then mlp3 over nodes -----------------
    edge2node_kernel<<<nodeRows, 128, 0, stream>>>(e_act, scale2, shift2, n_in_act);
    {
        StageArgs a = {};
        a.srcA = n_in_act;  // already BN'd + /V
        a.wp1 = pk3w1; a.wp2 = pk3w2; a.b1 = b1_3; a.b2 = b2_3;
        a.actOut = n_act; a.partial = partial;
        mlp_stage_kernel<128, 2><<<nodeRows / 64, 256, 0, stream>>>(a);
        bn_finalize_kernel<<<1, 128, 0, stream>>>(partial, nodeRows / 64,
                                                  1.0f / nodeRows, g_3, be_3, scale3, shift3);
    }
    // ---- 4) mlp4 over edges (gather BN3(n) x2 | BN2(e) skip) --------------
    {
        StageArgs a = {};
        a.srcA = n_act; a.scA = scale3; a.shA = shift3;
        a.srcSkip = e_act; a.scS = scale2; a.shS = shift2;
        a.send = send; a.recv = recv;
        a.wp1 = pk4w1; a.wp2 = pk4w2; a.b1 = b1_4; a.b2 = b2_4;
        a.actOut = e2_act; a.partial = partial;
        mlp_stage_kernel<384, 3><<<(int)(edgeRows / 64), 256, 0, stream>>>(a);
        bn_finalize_kernel<<<1, 128, 0, stream>>>(partial, (int)(edgeRows / 64),
                                                  1.0f / (float)edgeRows, g_4, be_4, scale4, shift4);
    }
    // ---- 5) fc + sigmoid --------------------------------------------------
    fc_sigmoid_kernel<<<(int)(edgeRows / 8), 256, 0, stream>>>(e2_act, scale4, shift4,
                                                               fcW, fcb, out);
}
